// MultiHeadAttention_74586402062628
// MI455X (gfx1250) — matrix operand order repaired and checked by reading, not compiled, not run on hardware
//
#include <hip/hip_runtime.h>
#include <hip/hip_bf16.h>
#include <math.h>

// ---------------------------------------------------------------------------
// MHA forward for MI455X (gfx1250, wave32, WMMA).
// Shapes: B=8, S=1000, E=1024, H=16, D=64.  All WMMA: f16 in, f32 accum.
// ---------------------------------------------------------------------------

#define BB   8
#define SS   1000
#define EE   1024
#define HH   16
#define DD   64
#define SP   1024              // padded seq length for V^T inner dim
#define MROWS (BB * SS)        // 8000 flat token rows

typedef _Float16 v16h __attribute__((ext_vector_type(16)));
typedef _Float16 v8h  __attribute__((ext_vector_type(8)));
typedef float    v8f  __attribute__((ext_vector_type(8)));

#define NEG_INF (-__builtin_inff())

#if __has_builtin(__builtin_amdgcn_sched_barrier)
#define SCHED_FENCE() __builtin_amdgcn_sched_barrier(0)
#else
#define SCHED_FENCE()
#endif

__device__ __forceinline__ v8f wmma_f16(v16h a, v16h b, v8f c) {
  // 8 args: (neg_a, A, neg_b, B, c_mod, C, reuse_a, reuse_b)
  return __builtin_amdgcn_wmma_f32_16x16x32_f16(false, a, false, b, (short)0, c,
                                                false, false);
}

union AFrag { v16h v; v8h h[2]; };

// A fragment (16x32, M x K), row-major source, contiguous layout:
// lane L: M = L%16; kbase = (L/16)*16; halves0-15 -> K=kbase+0..15.
__device__ __forceinline__ v16h load_a_frag(const _Float16* row_ptr, int kk,
                                            int kb) {
  AFrag u;
  u.h[0] = *(const v8h*)(row_ptr + kk + 2 * kb);
  u.h[1] = *(const v8h*)(row_ptr + kk + 2 * kb + 8);
  return u.v;
}

// B fragment (32x16, K x N) from a K-major (transposed) weight matrix with
// leading dim EE: lane L holds column n = n0 + L%16, halves are 16 contiguous
// K values starting at koff (= kk + (L/16)*16).
__device__ __forceinline__ v16h load_b_frag(const _Float16* w, int n, int koff) {
  return *(const v16h*)(w + (size_t)n * EE + koff);
}

// ---------------------------------------------------------------------------
// K1: conversions.  x -> f16; wq/wk/wv -> f16 transposed [3][H][D][E];
//     wp -> f16 transposed [F][E].
// ---------------------------------------------------------------------------
__global__ __launch_bounds__(256) void cvt_x(const float* __restrict__ x,
                                             _Float16* __restrict__ xh) {
  size_t i = (size_t)blockIdx.x * blockDim.x + threadIdx.x;
  xh[i] = (_Float16)x[i];
}

__global__ __launch_bounds__(256) void cvt_wqkv(const float* __restrict__ wq,
                                                const float* __restrict__ wk,
                                                const float* __restrict__ wv,
                                                _Float16* __restrict__ wT) {
  size_t i = (size_t)blockIdx.x * blockDim.x + threadIdx.x;
  const size_t HED = (size_t)HH * EE * DD;
  int which = (int)(i / HED);
  size_t hed = i - (size_t)which * HED;
  int h = (int)(hed / (EE * DD));
  int ed = (int)(hed % (EE * DD));
  int e = ed / DD;
  int d = ed - e * DD;
  const float* src = (which == 0) ? wq : (which == 1) ? wk : wv;
  // out layout: [(which*H + h)][d][e]  (K-major for B fragments)
  wT[(((size_t)which * HH + h) * DD + d) * EE + e] = (_Float16)src[hed];
}

__global__ __launch_bounds__(256) void cvt_wp(const float* __restrict__ wp,
                                              _Float16* __restrict__ wpT) {
  size_t i = (size_t)blockIdx.x * blockDim.x + threadIdx.x;
  int e = (int)(i / EE);
  int f = (int)(i % EE);
  wpT[(size_t)f * EE + e] = (_Float16)wp[i];
}

// ---------------------------------------------------------------------------
// K2: QKV projection GEMM.  M=8000, K=1024, N=64 per (which, head).
// Workgroup tile 128x64; 8 waves, each wave 16x64 (4 accumulators).
// k-loop unrolled 2x with ping-pong fragment sets; sched fences keep each
// refill AFTER the WMMA block that kills the old set, so the allocator can
// reload in place (no rotation movs) while loads still get a full WMMA
// block + backedge of latency cover.
// Q gets bias + 1/sqrt(S) folded in; V is stored transposed [d][s].
// ---------------------------------------------------------------------------
__global__ __launch_bounds__(256) void qkv_gemm(
    const _Float16* __restrict__ xh, const _Float16* __restrict__ wqkvT,
    const float* __restrict__ bq, const float* __restrict__ bk,
    const float* __restrict__ bv, _Float16* __restrict__ q,
    _Float16* __restrict__ k, _Float16* __restrict__ vT) {
  const int lane = threadIdx.x & 31;
  const int wid  = threadIdx.x >> 5;
  const int which = blockIdx.y / HH;
  const int head  = blockIdx.y % HH;
  const _Float16* w = wqkvT + (size_t)blockIdx.y * DD * EE;
  const float* bias = (which == 0) ? bq : (which == 1) ? bk : bv;

  const int m0 = blockIdx.x * 128 + wid * 16;
  int arow = m0 + (lane & 15);
  if (arow > MROWS - 1) arow = MROWS - 1;
  const _Float16* aptr = xh + (size_t)arow * EE;
  const int kb = (lane >> 4) * 8;
  const int kb16 = (lane >> 4) * 16;
  const int col = lane & 15;

  v8f c[4] = {v8f{}, v8f{}, v8f{}, v8f{}};
  v16h a0, a1, b0[4], b1[4];

  // prologue: fragment sets for kk = 0 and kk = 32
  a0 = load_a_frag(aptr, 0, kb);
#pragma unroll
  for (int j = 0; j < 4; ++j) b0[j] = load_b_frag(w, j * 16 + col, kb16);
  a1 = load_a_frag(aptr, 32, kb);
#pragma unroll
  for (int j = 0; j < 4; ++j) b1[j] = load_b_frag(w, j * 16 + col, 32 + kb16);

  for (int kk = 0; kk < EE - 64; kk += 64) {
#pragma unroll
    for (int j = 0; j < 4; ++j) c[j] = wmma_f16(a0, b0[j], c[j]);
    SCHED_FENCE();
    // refill set0 for kk+64; overlaps set1 WMMAs below
    a0 = load_a_frag(aptr, kk + 64, kb);
#pragma unroll
    for (int j = 0; j < 4; ++j)
      b0[j] = load_b_frag(w, j * 16 + col, kk + 64 + kb16);
    SCHED_FENCE();
#pragma unroll
    for (int j = 0; j < 4; ++j) c[j] = wmma_f16(a1, b1[j], c[j]);
    SCHED_FENCE();
    // refill set1 for kk+96; overlaps next iteration's set0 WMMAs
    a1 = load_a_frag(aptr, kk + 96, kb);
#pragma unroll
    for (int j = 0; j < 4; ++j)
      b1[j] = load_b_frag(w, j * 16 + col, kk + 96 + kb16);
    SCHED_FENCE();
  }
#pragma unroll
  for (int j = 0; j < 4; ++j) c[j] = wmma_f16(a0, b0[j], c[j]);
#pragma unroll
  for (int j = 0; j < 4; ++j) c[j] = wmma_f16(a1, b1[j], c[j]);

  const float qscale = 0.031622776601683794f;  // 1/sqrt(1000)
  const int rowoff = 8 * (lane >> 4);

  int bArr[8], sArr[8];
#pragma unroll
  for (int r = 0; r < 8; ++r) {
    int gr = m0 + rowoff + r;
    int b_ = gr / SS;
    bArr[r] = b_;
    sArr[r] = gr - b_ * SS;
  }
#pragma unroll
  for (int j = 0; j < 4; ++j) {
    int d = j * 16 + col;
    float bi = bias[head * DD + d];
#pragma unroll
    for (int r = 0; r < 8; ++r) {
      int gr = m0 + rowoff + r;
      if (gr >= MROWS) continue;
      float v = c[j][r] + bi;
      size_t bh = (size_t)bArr[r] * HH + head;
      if (which == 0)
        q[(bh * SS + sArr[r]) * DD + d] = (_Float16)(v * qscale);
      else if (which == 1)
        k[(bh * SS + sArr[r]) * DD + d] = (_Float16)v;
      else
        vT[(bh * DD + d) * SP + sArr[r]] = (_Float16)v;
    }
  }
}

// ---------------------------------------------------------------------------
// K3: causal flash attention per (b,h).  8 waves x 16 query rows = 128 rows
// per workgroup.  Key tiles of 64.  Scores: Q(16x64) @ K^T via WMMA; online
// softmax with half-wave shuffles; P transposed through per-wave LDS region
// (s_wait_dscnt, no barrier); P @ V from V^T fragments.  Sched fences keep
// the V loads issued right after the score WMMAs so the softmax VALU block
// hides their latency.
// ---------------------------------------------------------------------------
__global__ __launch_bounds__(256) void attn(const _Float16* __restrict__ q,
                                            const _Float16* __restrict__ k,
                                            const _Float16* __restrict__ vT,
                                            _Float16* __restrict__ y) {
  __shared__ _Float16 lds_p[8][16][64];   // 16 KB, per-wave private regions

  const int lane = threadIdx.x & 31;
  const int wid  = threadIdx.x >> 5;
  const int bh = blockIdx.y;
  const int b_ = bh / HH;
  const int h  = bh % HH;
  const int wq0 = blockIdx.x * 128 + wid * 16;
  const int kb = (lane >> 4) * 8;
  const int kb16 = (lane >> 4) * 16;
  const int col = lane & 15;
  const int rowoff = 8 * (lane >> 4);

  // Q fragments (two k-steps covering D=64); clamp padded rows
  int srow = wq0 + col;
  if (srow > SS - 1) srow = SS - 1;
  const _Float16* qrow = q + ((size_t)bh * SS + srow) * DD;
  v16h qa0 = load_a_frag(qrow, 0, kb);
  v16h qa1 = load_a_frag(qrow, 32, kb);

  const _Float16* kbase = k + (size_t)bh * SS * DD;
  const _Float16* vbase = vT + (size_t)bh * DD * SP;

  v8f o[4] = {v8f{}, v8f{}, v8f{}, v8f{}};
  float m_old[8], lsum[8];
#pragma unroll
  for (int r = 0; r < 8; ++r) { m_old[r] = NEG_INF; lsum[r] = 0.f; }

  const int smax = wq0 + 15;  // causal extent of this wave
  for (int t0 = 0; t0 <= smax; t0 += 64) {
    // ---- K fragments: batch all 8 loads, then 8 score WMMAs ----
    v16h kf0[4], kf1[4];
#pragma unroll
    for (int j = 0; j < 4; ++j) {
      int t = t0 + j * 16 + col;
      if (t > SS - 1) t = SS - 1;                 // masked below anyway
      const _Float16* kr = kbase + (size_t)t * DD;
      kf0[j] = *(const v16h*)(kr + kb16);         // K-dim = d, contiguous
      kf1[j] = *(const v16h*)(kr + 32 + kb16);
    }
    v8f sc[4];
#pragma unroll
    for (int j = 0; j < 4; ++j) {
      v8f z = {};
      z = wmma_f16(qa0, kf0[j], z);
      z = wmma_f16(qa1, kf1[j], z);
      sc[j] = z;
    }
    SCHED_FENCE();
    // ---- issue V fragment loads now; softmax VALU below hides them ----
    v16h vf0[4], vf1[4];
#pragma unroll
    for (int j = 0; j < 4; ++j) {
      const _Float16* vr = vbase + (size_t)(j * 16 + col) * SP + t0;
      vf0[j] = *(const v16h*)(vr + kb16);         // K-dim = t, contiguous
      vf1[j] = *(const v16h*)(vr + 32 + kb16);
    }
    SCHED_FENCE();
    // ---- causal mask ----
#pragma unroll
    for (int j = 0; j < 4; ++j)
#pragma unroll
      for (int r = 0; r < 8; ++r) {
        int t = t0 + j * 16 + col;
        int s = wq0 + rowoff + r;
        if (t > s) sc[j][r] = NEG_INF;
      }
    // ---- row max (reduce across the 16-lane half-group) ----
    float mnew[8], alpha[8];
#pragma unroll
    for (int r = 0; r < 8; ++r) {
      float m = fmaxf(fmaxf(sc[0][r], sc[1][r]), fmaxf(sc[2][r], sc[3][r]));
#pragma unroll
      for (int sh = 1; sh < 16; sh <<= 1) m = fmaxf(m, __shfl_xor(m, sh, 32));
      mnew[r] = fmaxf(m_old[r], m);
      alpha[r] = (m_old[r] == NEG_INF) ? 0.f : __expf(m_old[r] - mnew[r]);
      m_old[r] = mnew[r];
    }
    // ---- rescale O, exponentiate, stash P to LDS, row sums ----
    float psum[8];
#pragma unroll
    for (int r = 0; r < 8; ++r) psum[r] = 0.f;
#pragma unroll
    for (int j = 0; j < 4; ++j) {
#pragma unroll
      for (int r = 0; r < 8; ++r) {
        o[j][r] *= alpha[r];
        float p = __expf(sc[j][r] - mnew[r]);   // exp(-inf - m) == 0
        psum[r] += p;
        lds_p[wid][rowoff + r][j * 16 + col] = (_Float16)p;
      }
    }
#pragma unroll
    for (int r = 0; r < 8; ++r) {
      float s = psum[r];
#pragma unroll
      for (int sh = 1; sh < 16; sh <<= 1) s += __shfl_xor(s, sh, 32);
      lsum[r] = lsum[r] * alpha[r] + s;
    }
    // per-wave LDS store -> load ordering (DS ops are in-order per wave;
    // explicit split-counter wait, no workgroup barrier needed)
    asm volatile("s_wait_dscnt 0" ::: "memory");
    // ---- P as A fragments (transpose via LDS) ----
    const _Float16* prow = &lds_p[wid][col][0];
    v16h pa0 = load_a_frag(prow, 0, kb);
    v16h pa1 = load_a_frag(prow, 32, kb);
    // ---- O += P @ V ----
#pragma unroll
    for (int j = 0; j < 4; ++j) {
      o[j] = wmma_f16(pa0, vf0[j], o[j]);
      o[j] = wmma_f16(pa1, vf1[j], o[j]);
    }
  }

  // ---- normalize and write y[b*S+s][h*64+d] (concat heads) ----
#pragma unroll
  for (int j = 0; j < 4; ++j)
#pragma unroll
    for (int r = 0; r < 8; ++r) {
      int s = wq0 + rowoff + r;
      if (s < SS) {
        float val = o[j][r] / lsum[r];
        y[((size_t)b_ * SS + s) * EE + h * DD + j * 16 + col] = (_Float16)val;
      }
    }
}

// ---------------------------------------------------------------------------
// K4: output projection + exact GELU.  M=8000, N=1024, K=1024.
// Workgroup tile 128x128; each wave 16x128 (8 accumulators).  The two
// 4-fragment B groups are software-pipelined against each other; sched
// fences keep each group's refill after the WMMA block that kills it.
// ---------------------------------------------------------------------------
__global__ __launch_bounds__(256) void out_proj(const _Float16* __restrict__ y,
                                                const _Float16* __restrict__ wpT,
                                                const float* __restrict__ bp,
                                                float* __restrict__ out) {
  const int lane = threadIdx.x & 31;
  const int wid  = threadIdx.x >> 5;
  const int m0 = blockIdx.x * 128 + wid * 16;
  const int n0 = blockIdx.y * 128;
  const int kb = (lane >> 4) * 8;
  const int kb16 = (lane >> 4) * 16;
  const int col = lane & 15;
  const int rowoff = 8 * (lane >> 4);

  int arow = m0 + col;
  if (arow > MROWS - 1) arow = MROWS - 1;
  const _Float16* aptr = y + (size_t)arow * EE;

  v8f c[8] = {v8f{}, v8f{}, v8f{}, v8f{}, v8f{}, v8f{}, v8f{}, v8f{}};

  // prologue: A for kk=0, B group0 for kk=0
  v16h a_cur = load_a_frag(aptr, 0, kb);
  v16h bA[4];
#pragma unroll
  for (int j = 0; j < 4; ++j)
    bA[j] = load_b_frag(wpT, n0 + j * 16 + col, kb16);

  for (int kk = 0; kk < EE - 32; kk += 32) {
    // B group1(kk) + A(kk+32): overlap group0 WMMAs
    v16h bB[4];
#pragma unroll
    for (int j = 0; j < 4; ++j)
      bB[j] = load_b_frag(wpT, n0 + (j + 4) * 16 + col, kk + kb16);
    v16h a_nxt = load_a_frag(aptr, kk + 32, kb);
    SCHED_FENCE();
#pragma unroll
    for (int j = 0; j < 4; ++j) c[j] = wmma_f16(a_cur, bA[j], c[j]);
    SCHED_FENCE();
    // B group0(kk+32): overlap group1 WMMAs
#pragma unroll
    for (int j = 0; j < 4; ++j)
      bA[j] = load_b_frag(wpT, n0 + j * 16 + col, kk + 32 + kb16);
    SCHED_FENCE();
#pragma unroll
    for (int j = 0; j < 4; ++j) c[j + 4] = wmma_f16(a_cur, bB[j], c[j + 4]);
    SCHED_FENCE();
    a_cur = a_nxt;
  }
  {
    v16h bB[4];
#pragma unroll
    for (int j = 0; j < 4; ++j)
      bB[j] = load_b_frag(wpT, n0 + (j + 4) * 16 + col, EE - 32 + kb16);
#pragma unroll
    for (int j = 0; j < 4; ++j) c[j] = wmma_f16(a_cur, bA[j], c[j]);
#pragma unroll
    for (int j = 0; j < 4; ++j) c[j + 4] = wmma_f16(a_cur, bB[j], c[j + 4]);
  }

#pragma unroll
  for (int nf = 0; nf < 8; ++nf) {
    int f = n0 + nf * 16 + col;
    float bi = bp[f];
#pragma unroll
    for (int r = 0; r < 8; ++r) {
      int gr = m0 + rowoff + r;
      if (gr < MROWS) {
        float xv = c[nf][r] + bi;
        float g = 0.5f * xv * (1.0f + erff(xv * 0.70710678118654752f));
        out[(size_t)gr * EE + f] = g;
      }
    }
  }
}

// ---------------------------------------------------------------------------
// Host launcher
// ---------------------------------------------------------------------------
extern "C" void kernel_launch(void* const* d_in, const int* in_sizes, int n_in,
                              void* d_out, int out_size, void* d_ws,
                              size_t ws_size, hipStream_t stream) {
  const float* x  = (const float*)d_in[0];
  const float* wq = (const float*)d_in[1];
  const float* bq = (const float*)d_in[2];
  const float* wk = (const float*)d_in[3];
  const float* bk = (const float*)d_in[4];
  const float* wv = (const float*)d_in[5];
  const float* bv = (const float*)d_in[6];
  const float* wp = (const float*)d_in[7];
  const float* bp = (const float*)d_in[8];
  float* out = (float*)d_out;

  // workspace carve-out (~87 MB total)
  char* ws = (char*)d_ws;
  size_t off = 0;
  auto carve = [&](size_t bytes) -> void* {
    void* p = ws + off;
    off = (off + bytes + 255) & ~(size_t)255;
    return p;
  };
  _Float16* xh     = (_Float16*)carve((size_t)MROWS * EE * 2);        // 16.4 MB
  _Float16* wqkvT  = (_Float16*)carve((size_t)3 * HH * DD * EE * 2);  //  6.3 MB
  _Float16* wpT    = (_Float16*)carve((size_t)EE * EE * 2);           //  2.1 MB
  _Float16* qh     = (_Float16*)carve((size_t)BB * HH * SS * DD * 2); // 16.4 MB
  _Float16* kh     = (_Float16*)carve((size_t)BB * HH * SS * DD * 2); // 16.4 MB
  _Float16* vTh    = (_Float16*)carve((size_t)BB * HH * DD * SP * 2); // 16.8 MB
  _Float16* yh     = (_Float16*)carve((size_t)MROWS * EE * 2);        // 16.4 MB
  (void)ws_size; (void)in_sizes; (void)n_in; (void)out_size;

  // K1: conversions / transposes
  cvt_x<<<(MROWS * EE) / 256, 256, 0, stream>>>(x, xh);
  cvt_wqkv<<<(3 * HH * EE * DD) / 256, 256, 0, stream>>>(wq, wk, wv, wqkvT);
  cvt_wp<<<(EE * EE) / 256, 256, 0, stream>>>(wp, wpT);

  // K2: QKV projections (63 M-tiles of 128 rows; y = which*H + head)
  qkv_gemm<<<dim3((MROWS + 127) / 128, 3 * HH), 256, 0, stream>>>(
      xh, wqkvT, bq, bk, bv, qh, kh, vTh);

  // K3: causal flash attention (8 query tiles of 128; one block per (tile, b,h))
  attn<<<dim3(8, BB * HH), 256, 0, stream>>>(qh, kh, vTh, yh);

  // K4: output projection + GELU
  out_proj<<<dim3((MROWS + 127) / 128, EE / 128), 256, 0, stream>>>(yh, wpT, bp,
                                                                    out);
}
